// Stress2Fringe_17892833755680
// MI455X (gfx1250) — matrix operand order repaired and checked by reading, not compiled, not run on hardware
//
#include <hip/hip_runtime.h>

typedef __attribute__((ext_vector_type(16))) _Float16 v16h;
typedef __attribute__((ext_vector_type(8)))  float    v8f;

#define B_      8
#define P_      (1024*1024)
#define NL      371
#define NK      384            // K padded to 12 chunks of 32
#define NCHUNK  12
#define COEF    2.1991148575128552e-4f   // 2*pi*H*C/1e-9
#define SMAG    72000000.0f
#define TILES_PER_BATCH 8192   // 1M pixels / 128 pixels per block

// -------- workspace layout (byte offsets) --------
#define WS_MM     0            // [8][256][2] f32  partial min/max
#define WS_BATCH  16384        // [8][2]      f32  {mn, 1/(mx-mn)}
#define WS_AW     16448        // [384]       f32  coef/lambda (0 padded)
#define WS_AFRAG  32768        // [12][32][16] f16 WMMA A fragments
#define WS_BMAX   65536        // [8][8192]   f32  per-block iso max
#define WS_BINV   327680       // [8]         f32  1/batch iso max

// Build coef/lambda table and A-matrix fragments in WMMA 16-bit A layout:
// lanes 0-15 : row M=lane,    elems 0..7 -> K=0..7,  elems 8..15 -> K=16..23
// lanes 16-31: row M=lane-16, elems 0..7 -> K=8..15, elems 8..15 -> K=24..31
__global__ void k_pack(const float* __restrict__ ssi,   // [371][3]
                       float* __restrict__ aw,
                       _Float16* __restrict__ afrag) {
  for (int k = threadIdx.x; k < NK; k += blockDim.x)
    aw[k] = (k < NL) ? (COEF / (390.0f + (float)k)) : 0.0f;
  for (int t = threadIdx.x; t < NCHUNK * 32 * 16; t += blockDim.x) {
    int e    = t & 15;
    int lane = (t >> 4) & 31;
    int kc   = t >> 9;
    int m    = lane & 15;
    int K;
    if (lane < 16) K = kc * 32 + (e < 8 ? e     : e + 8);
    else           K = kc * 32 + (e < 8 ? e + 8 : e + 16);
    float v = (K < NL && m < 3) ? ssi[K * 3 + m] : 0.0f;
    afrag[t] = (_Float16)v;
  }
}

__global__ void __launch_bounds__(256)
k_minmax_partial(const float* __restrict__ stress, float* __restrict__ mm) {
  const int b = blockIdx.y, t = threadIdx.x;
  float lo = 3.4e38f, hi = -3.4e38f;
  int base = b * P_ + blockIdx.x * 256 + t;
  #pragma unroll
  for (int i = 0; i < 16; ++i) {
    float v = stress[base + i * 65536];
    lo = fminf(lo, v); hi = fmaxf(hi, v);
  }
  __shared__ float slo[256], shi[256];
  slo[t] = lo; shi[t] = hi;
  __syncthreads();
  for (int off = 128; off > 0; off >>= 1) {
    if (t < off) { slo[t] = fminf(slo[t], slo[t + off]);
                   shi[t] = fmaxf(shi[t], shi[t + off]); }
    __syncthreads();
  }
  if (t == 0) { mm[(b * 256 + blockIdx.x) * 2 + 0] = slo[0];
                mm[(b * 256 + blockIdx.x) * 2 + 1] = shi[0]; }
}

__global__ void __launch_bounds__(256)
k_minmax_final(const float* __restrict__ mm, float* __restrict__ batchp) {
  const int b = blockIdx.x, t = threadIdx.x;
  __shared__ float slo[256], shi[256];
  slo[t] = mm[(b * 256 + t) * 2 + 0];
  shi[t] = mm[(b * 256 + t) * 2 + 1];
  __syncthreads();
  for (int off = 128; off > 0; off >>= 1) {
    if (t < off) { slo[t] = fminf(slo[t], slo[t + off]);
                   shi[t] = fmaxf(shi[t], shi[t + off]); }
    __syncthreads();
  }
  if (t == 0) { batchp[2 * b + 0] = slo[0];
                batchp[2 * b + 1] = 1.0f / (shi[0] - slo[0]); }
}

// Main kernel: 256 threads = 8 waves; each wave handles 16 pixels x 384 K via
// 12 x v_wmma_f32_16x16x32_f16. B layout: lane<16 -> K=chunk+[0..7,16..23],
// lane>=16 -> K=chunk+[8..15,24..31]; column N = lane&15 (pixel within tile).
__global__ void __launch_bounds__(256)
k_main(const float* __restrict__ stress,
       const float* __restrict__ batchp,
       const float* __restrict__ aw,
       const v16h*  __restrict__ afrag,
       float* __restrict__ outIso,
       float* __restrict__ outNom,
       float* __restrict__ bmax) {
  const int b    = blockIdx.y;
  const int lane = threadIdx.x & 31;
  const int wv   = threadIdx.x >> 5;
  const int pix  = (blockIdx.x * 8 + wv) * 16 + (lane & 15);

  const float mn   = batchp[2 * b + 0];
  const float invr = batchp[2 * b + 1];

  float s  = stress[b * P_ + pix];
  float sn = (s - mn) * invr;
  float sr = sn * SMAG;
  if (lane < 16) outNom[b * P_ + pix] = sn;

  const int kshift = lane & 16;

  v8f acc = {};
  #pragma unroll 1
  for (int kc = 0; kc < NCHUNK; ++kc) {
    v16h a = afrag[kc * 32 + lane];
    const int kbase = kc * 32 + (kshift >> 1);
    v16h bt;
    #pragma unroll
    for (int e = 0; e < 16; ++e) {
      float phase = sr * aw[kbase + (e < 8 ? e : e + 8)];
      float term  = 0.5f - 0.5f * __cosf(phase);
      bt[e] = (_Float16)term;
    }
    acc = __builtin_amdgcn_wmma_f32_16x16x32_f16(false, a, false, bt,
                                                 (short)0, acc, false, false);
  }

  float m = 0.0f;
  if (lane < 16) {                 // D rows M=0,1,2 live in VGPRs 0..2, lanes 0-15
    float c0 = acc[0], c1 = acc[1], c2 = acc[2];
    outIso[((size_t)b * 3 + 0) * P_ + pix] = c0;
    outIso[((size_t)b * 3 + 1) * P_ + pix] = c1;
    outIso[((size_t)b * 3 + 2) * P_ + pix] = c2;
    m = fmaxf(c0, fmaxf(c1, c2));
  }
  __shared__ float smax[256];
  smax[threadIdx.x] = m;
  __syncthreads();
  for (int off = 128; off > 0; off >>= 1) {
    if (threadIdx.x < off)
      smax[threadIdx.x] = fmaxf(smax[threadIdx.x], smax[threadIdx.x + off]);
    __syncthreads();
  }
  if (threadIdx.x == 0) bmax[b * TILES_PER_BATCH + blockIdx.x] = smax[0];
}

__global__ void __launch_bounds__(256)
k_isomax_final(const float* __restrict__ bmax, float* __restrict__ binv) {
  const int b = blockIdx.x, t = threadIdx.x;
  float m = 0.0f;
  for (int j = t; j < TILES_PER_BATCH; j += 256)
    m = fmaxf(m, bmax[b * TILES_PER_BATCH + j]);
  __shared__ float smax[256];
  smax[t] = m;
  __syncthreads();
  for (int off = 128; off > 0; off >>= 1) {
    if (t < off) smax[t] = fmaxf(smax[t], smax[t + off]);
    __syncthreads();
  }
  if (t == 0) binv[b] = 1.0f / smax[0];
}

__global__ void __launch_bounds__(256)
k_normalize(float* __restrict__ iso, const float* __restrict__ binv, int n) {
  int stride = gridDim.x * blockDim.x;
  for (int i = blockIdx.x * blockDim.x + threadIdx.x; i < n; i += stride) {
    int b = i / (3 * P_);
    iso[i] *= binv[b];
  }
}

extern "C" void kernel_launch(void* const* d_in, const int* in_sizes, int n_in,
                              void* d_out, int out_size, void* d_ws, size_t ws_size,
                              hipStream_t stream) {
  const float* stress = (const float*)d_in[0];   // [8,1,1024,1024] f32
  const float* ssi    = (const float*)d_in[1];   // [371,3] f32

  float* out    = (float*)d_out;
  float* outIso = out;                           // [8,3,1024,1024]
  float* outNom = out + (size_t)B_ * 3 * P_;     // [8,1024,1024]

  char*  ws     = (char*)d_ws;
  float*    mm     = (float*)(ws + WS_MM);
  float*    batchp = (float*)(ws + WS_BATCH);
  float*    aw     = (float*)(ws + WS_AW);
  _Float16* afrag  = (_Float16*)(ws + WS_AFRAG);
  float*    bmax   = (float*)(ws + WS_BMAX);
  float*    binv   = (float*)(ws + WS_BINV);

  k_pack<<<1, 256, 0, stream>>>(ssi, aw, afrag);
  k_minmax_partial<<<dim3(256, B_), 256, 0, stream>>>(stress, mm);
  k_minmax_final<<<B_, 256, 0, stream>>>(mm, batchp);
  k_main<<<dim3(TILES_PER_BATCH, B_), 256, 0, stream>>>(
      stress, batchp, aw, (const v16h*)afrag, outIso, outNom, bmax);
  k_isomax_final<<<B_, 256, 0, stream>>>(bmax, binv);
  k_normalize<<<8192, 256, 0, stream>>>(outIso, binv, B_ * 3 * P_);
}
